// NodeEmbedderAggr_82506321756633
// MI455X (gfx1250) — matrix operand order repaired and checked by reading, not compiled, not run on hardware
//
#include <hip/hip_runtime.h>
#include <hip/hip_bf16.h>

// ---------------------------------------------------------------------------
// NodeEmbedderAggr for MI455X (gfx1250, wave32, WMMA f32_16x16x32_f16)
// ---------------------------------------------------------------------------

typedef __attribute__((ext_vector_type(16))) _Float16 v16h;
typedef __attribute__((ext_vector_type(8)))  _Float16 v8h;
typedef __attribute__((ext_vector_type(8)))  float    v8f;
typedef __attribute__((ext_vector_type(4)))  float    v4f;

#define RRELU_SLOPE 0.22916666666666667f

#define T_TOK   262144
#define N_NODES 8192
#define BT      128      // tokens per block
#define NBLK    (T_TOK / BT)   // 2048 blocks per stream
#define XSTR    168      // 160 K-cols (136 real + pad) + 8 pad halves
#define HSTR    264      // 256 + 8 pad halves
#define YSTR    132      // 128 + 4 pad floats
#define CSTR    392      // 384 + 8 pad halves (combine)

// workspace byte offsets
#define OFF_SF   0u
#define OFF_SR   4194304u
#define OFF_SM   8388608u
#define OFF_CF   12582912u
#define OFF_CR   12615680u
#define OFF_CM   12648448u
#define OFF_W1F  12681216u   // 3 x 81920 B  (KT=5,  NT=16)
#define OFF_W2F  12926976u   // 3 x 65536 B  (KT=8,  NT=8)
#define OFF_WCF  13123584u   // 98304 B      (KT=12, NT=8)

union V16U { v16h v; v8h h[2]; };

__device__ __forceinline__ float rrelu(float x) {
    // max/min/fma form: 3 independent VALU ops, no VCC/SGPR dependency chain
    return fmaxf(x, 0.f) + RRELU_SLOPE * fminf(x, 0.f);
}

// B-fragment (K=32 x N=16) pre-swizzled in global: 512 halves, lane-major.
__device__ __forceinline__ v16h load_frag_g(const _Float16* base, int frag, int lane) {
    const v8h* p = (const v8h*)(base + (size_t)frag * 512 + lane * 16);
    V16U u; u.h[0] = p[0]; u.h[1] = p[1];
    return u.v;
}

// A-fragment (M=16 x K=32) from an LDS row (row = one matrix row, f16).
// lanes 0-15: K = 0..7 / 16..23 ; lanes 16-31: K = 8..15 / 24..31
__device__ __forceinline__ v16h load_frag_lds(const _Float16* row, int kt, int koff) {
    V16U u;
    u.h[0] = *(const v8h*)(row + kt * 32 + koff);
    u.h[1] = *(const v8h*)(row + kt * 32 + koff + 16);
    return u.v;
}

__device__ __forceinline__ v8f wmma_f16(v16h a, v16h b, v8f c) {
    return __builtin_amdgcn_wmma_f32_16x16x32_f16(false, a, false, b, (short)0, c,
                                                  false, false);
}

// ---------------------------------------------------------------------------
// zero workspace region
// ---------------------------------------------------------------------------
__global__ void zero_kernel(float* p, unsigned n) {
    unsigned i = blockIdx.x * 256u + threadIdx.x;
    if (i < n) p[i] = 0.f;
}

// ---------------------------------------------------------------------------
// pack a row-major f32 weight [realK x Ncols] into f16 B-fragments
// frag = kt*NT + nt ; lane l, half i -> K = kt*32 + (l/16)*8 + (i%8) + (i/8)*16, N = nt*16 + l%16
// ---------------------------------------------------------------------------
__global__ void pack_frag_kernel(const float* __restrict__ W, _Float16* __restrict__ dst,
                                 int realK, int Ncols, int NT) {
    int idx  = blockIdx.x * 256 + threadIdx.x;
    int frag = idx >> 9;
    int r    = idx & 511;
    int lane = r >> 4, i = r & 15;
    int kt = frag / NT, nt = frag % NT;
    int K = kt * 32 + (lane >> 4) * 8 + (i & 7) + (i >> 3) * 16;
    int N = nt * 16 + (lane & 15);
    float v = (K < realK) ? W[(size_t)K * Ncols + N] : 0.f;
    dst[idx] = (_Float16)v;
}

// ---------------------------------------------------------------------------
// fused per-stream: t2v -> [t2v|X] @ W1 -> rrelu -> @ W2 -> rrelu ->
// run-length segmented reduction (seg sorted) -> global f32 atomics
// 256 threads = 8 waves, 128 tokens/block; all 3 streams in one launch
// ---------------------------------------------------------------------------
struct StreamArgs {
    const float* t; const float* feats; const int* seg;
    const _Float16* W1F; const float* b1;
    const _Float16* W2F; const float* b2;
    float* S; float* CNT;
};
struct MlpArgs {
    StreamArgs st[3];
    const float* Wt; const float* bt;
};

__global__ __launch_bounds__(256) void mlp_aggr_kernel(MlpArgs args) {
    extern __shared__ char smem[];
    _Float16* sX   = (_Float16*)smem;                 // 128 x XSTR f16 (43008 B)
    float*    sY   = (float*)smem;                    // aliases sX: 128 x YSTR f32 (67584 B)
    _Float16* sH   = (_Float16*)(smem + 67584);       // 128 x HSTR f16 (67584 B)
    int*      sSeg = (int*)(smem + 135168);           // 128 ints

    const int sid  = blockIdx.x >> 11;                // stream id (uniform)
    const StreamArgs& A = args.st[sid];
    const float* __restrict__ t     = A.t;
    const float* __restrict__ feats = A.feats;
    const int*   __restrict__ seg   = A.seg;
    const _Float16* __restrict__ W1F = A.W1F;
    const float* __restrict__ b1    = A.b1;
    const _Float16* __restrict__ W2F = A.W2F;
    const float* __restrict__ b2    = A.b2;
    float* __restrict__ S   = A.S;
    float* __restrict__ CNT = A.CNT;
    const float* __restrict__ Wt = args.Wt;
    const float* __restrict__ bt = args.bt;

    const int tid  = threadIdx.x;
    const int tok0 = (blockIdx.x & (NBLK - 1)) * BT;

    // ---- phase A: stage activations (t2v + features) as f16 in LDS ----
    if (tid < BT) {
        int tok = tok0 + tid;
        sSeg[tid] = seg[tok];
        float tv = t[tok];
#pragma unroll
        for (int j = 0; j < 8; ++j) {
            float x = tv * Wt[j] + bt[j];
            sX[tid * XSTR + j] = (_Float16)(j == 0 ? x : __sinf(x));
        }
#pragma unroll
        for (int j = 136; j < 160; ++j) sX[tid * XSTR + j] = (_Float16)0.f;
    }
    {
        const v4f* f4 = (const v4f*)feats;            // 32 x v4f per token row
#pragma unroll
        for (int it = 0; it < 16; ++it) {
            int idx = it * 256 + tid;                 // 4096 v4f per block
            int row = idx >> 5, c4 = idx & 31;
            v4f v = f4[(size_t)tok0 * 32 + idx];
            _Float16* p = sX + row * XSTR + 8 + c4 * 4;
            p[0] = (_Float16)v.x; p[1] = (_Float16)v.y;
            p[2] = (_Float16)v.z; p[3] = (_Float16)v.w;
        }
    }
    __syncthreads();

    const int w    = tid >> 5;
    const int lane = tid & 31;
    const int l16  = lane & 15;
    const int koff = (lane >> 4) * 8;   // A-frag K sub-offset
    const int hi8  = (lane >> 4) * 8;   // C/D row offset

    // ---- layer 1: [16 x 160] @ [160 x 256] per wave ----
    v16h a1[5];
    {
        const _Float16* xrow = sX + (w * 16 + l16) * XSTR;
#pragma unroll
        for (int kt = 0; kt < 5; ++kt) a1[kt] = load_frag_lds(xrow, kt, koff);
    }
#pragma unroll
    for (int nt = 0; nt < 16; ++nt) {
        float bv = b1[nt * 16 + l16];
        v8f acc = {bv, bv, bv, bv, bv, bv, bv, bv};
#pragma unroll
        for (int kt = 0; kt < 5; ++kt) {
            v16h bm = load_frag_g(W1F, kt * 16 + nt, lane);
            acc = wmma_f16(a1[kt], bm, acc);
        }
#pragma unroll
        for (int r = 0; r < 8; ++r)
            sH[(w * 16 + r + hi8) * HSTR + nt * 16 + l16] = (_Float16)rrelu(acc[r]);
    }
    __syncthreads();   // sX dead from here; sY may overwrite it

    // ---- layer 2: [16 x 256] @ [256 x 128] per wave ----
    v16h a2[8];
    {
        const _Float16* hrow = sH + (w * 16 + l16) * HSTR;
#pragma unroll
        for (int kt = 0; kt < 8; ++kt) a2[kt] = load_frag_lds(hrow, kt, koff);
    }
#pragma unroll
    for (int nt = 0; nt < 8; ++nt) {
        float bv = b2[nt * 16 + l16];
        v8f acc = {bv, bv, bv, bv, bv, bv, bv, bv};
#pragma unroll
        for (int kt = 0; kt < 8; ++kt) {
            v16h bm = load_frag_g(W2F, kt * 8 + nt, lane);
            acc = wmma_f16(a2[kt], bm, acc);
        }
#pragma unroll
        for (int r = 0; r < 8; ++r)
            sY[(w * 16 + r + hi8) * YSTR + nt * 16 + l16] = rrelu(acc[r]);
    }
    __syncthreads();

    // ---- run-length segmented sum (seg sorted) -> global atomics ----
    {
        int col = tid & 127;
        int r0  = (tid >> 7) * 64;
        int cur = sSeg[r0];
        float acc = sY[r0 * YSTR + col];
        for (int rr = r0 + 1; rr < r0 + 64; ++rr) {
            int   sg = sSeg[rr];
            float v  = sY[rr * YSTR + col];
            if (sg != cur) {
                atomicAdd(&S[(size_t)cur * 128 + col], acc);
                cur = sg; acc = v;
            } else {
                acc += v;
            }
        }
        atomicAdd(&S[(size_t)cur * 128 + col], acc);
    }
    if (tid < BT) {
        bool start = (tid == 0) || (sSeg[tid] != sSeg[tid - 1]);
        if (start) {
            int sg = sSeg[tid];
            int len = 1;
            for (int rr = tid + 1; rr < BT && sSeg[rr] == sg; ++rr) ++len;
            atomicAdd(&CNT[sg], (float)len);
        }
    }
}

// ---------------------------------------------------------------------------
// combine: out = rrelu(concat(Sf/c, Sr/c, Sm/c) @ Wc + bc), 128 nodes/block
// ---------------------------------------------------------------------------
__global__ __launch_bounds__(256) void combine_kernel(
    const float* __restrict__ Sf, const float* __restrict__ Sr, const float* __restrict__ Sm,
    const float* __restrict__ Cf, const float* __restrict__ Cr, const float* __restrict__ Cm,
    const _Float16* __restrict__ WcF, const float* __restrict__ bc,
    float* __restrict__ out) {

    extern __shared__ char smem[];
    _Float16* sX = (_Float16*)smem;   // 128 x CSTR f16 (100352 B)

    const int tid = threadIdx.x;
    const int n0  = blockIdx.x * 128;

    const float* Sarr[3] = {Sf, Sr, Sm};
    const float* Carr[3] = {Cf, Cr, Cm};
#pragma unroll
    for (int s = 0; s < 3; ++s) {
        const v4f* p4 = (const v4f*)Sarr[s];
        const float* cp = Carr[s];
#pragma unroll
        for (int it = 0; it < 4; ++it) {
            int idx  = it * 256 + tid;              // 1024 v4f per stream
            int node = idx >> 5, c4 = idx & 31;
            v4f v = p4[(size_t)n0 * 32 + idx];
            float c = cp[n0 + node];
            float inv = 1.f / (c > 1.f ? c : 1.f);
            _Float16* q = sX + node * CSTR + s * 128 + c4 * 4;
            q[0] = (_Float16)(v.x * inv); q[1] = (_Float16)(v.y * inv);
            q[2] = (_Float16)(v.z * inv); q[3] = (_Float16)(v.w * inv);
        }
    }
    __syncthreads();

    const int w    = tid >> 5;
    const int lane = tid & 31;
    const int l16  = lane & 15;
    const int koff = (lane >> 4) * 8;
    const int hi8  = koff;

    v16h a[12];
    {
        const _Float16* xrow = sX + (w * 16 + l16) * CSTR;
#pragma unroll
        for (int kt = 0; kt < 12; ++kt) a[kt] = load_frag_lds(xrow, kt, koff);
    }
#pragma unroll
    for (int nt = 0; nt < 8; ++nt) {
        float bv = bc[nt * 16 + l16];
        v8f acc = {bv, bv, bv, bv, bv, bv, bv, bv};
#pragma unroll
        for (int kt = 0; kt < 12; ++kt) {
            v16h bm = load_frag_g(WcF, kt * 8 + nt, lane);
            acc = wmma_f16(a[kt], bm, acc);
        }
#pragma unroll
        for (int r = 0; r < 8; ++r)
            out[(size_t)(n0 + w * 16 + r + hi8) * 128 + nt * 16 + l16] = rrelu(acc[r]);
    }
}

// ---------------------------------------------------------------------------
extern "C" void kernel_launch(void* const* d_in, const int* in_sizes, int n_in,
                              void* d_out, int out_size, void* d_ws, size_t ws_size,
                              hipStream_t stream) {
    (void)in_sizes; (void)n_in; (void)out_size; (void)ws_size;

    const float* t_f   = (const float*)d_in[0];
    const float* f_in  = (const float*)d_in[1];
    const int*   seg_f = (const int*)d_in[2];
    const float* t_r   = (const float*)d_in[3];
    const float* r_in  = (const float*)d_in[4];
    const int*   seg_r = (const int*)d_in[5];
    const float* t_m   = (const float*)d_in[6];
    const float* m_in  = (const float*)d_in[7];
    const int*   seg_m = (const int*)d_in[8];
    // d_in[9] = n_nodes (compile-time N_NODES)
    const float* Wt  = (const float*)d_in[10];
    const float* bt  = (const float*)d_in[11];
    const float* Wf1 = (const float*)d_in[12]; const float* bf1 = (const float*)d_in[13];
    const float* Wf2 = (const float*)d_in[14]; const float* bf2 = (const float*)d_in[15];
    const float* Wr1 = (const float*)d_in[16]; const float* br1 = (const float*)d_in[17];
    const float* Wr2 = (const float*)d_in[18]; const float* br2 = (const float*)d_in[19];
    const float* Wm1 = (const float*)d_in[20]; const float* bm1 = (const float*)d_in[21];
    const float* Wm2 = (const float*)d_in[22]; const float* bm2 = (const float*)d_in[23];
    const float* Wc  = (const float*)d_in[24]; const float* bc  = (const float*)d_in[25];

    char* ws = (char*)d_ws;
    float* Sf = (float*)(ws + OFF_SF);
    float* Sr = (float*)(ws + OFF_SR);
    float* Sm = (float*)(ws + OFF_SM);
    float* Cf = (float*)(ws + OFF_CF);
    float* Cr = (float*)(ws + OFF_CR);
    float* Cm = (float*)(ws + OFF_CM);
    _Float16* W1Ff = (_Float16*)(ws + OFF_W1F);
    _Float16* W1Fr = (_Float16*)(ws + OFF_W1F + 81920u);
    _Float16* W1Fm = (_Float16*)(ws + OFF_W1F + 163840u);
    _Float16* W2Ff = (_Float16*)(ws + OFF_W2F);
    _Float16* W2Fr = (_Float16*)(ws + OFF_W2F + 65536u);
    _Float16* W2Fm = (_Float16*)(ws + OFF_W2F + 131072u);
    _Float16* WCF  = (_Float16*)(ws + OFF_WCF);

    // 1) zero segment sums + counts
    unsigned nz = OFF_W1F / 4u;   // floats in [0, OFF_W1F)
    zero_kernel<<<(nz + 255u) / 256u, 256, 0, stream>>>((float*)ws, nz);

    // 2) pack weights into f16 B-fragment order
    pack_frag_kernel<<<160, 256, 0, stream>>>(Wf1, W1Ff, 136, 256, 16);
    pack_frag_kernel<<<160, 256, 0, stream>>>(Wr1, W1Fr, 136, 256, 16);
    pack_frag_kernel<<<160, 256, 0, stream>>>(Wm1, W1Fm, 136, 256, 16);
    pack_frag_kernel<<<128, 256, 0, stream>>>(Wf2, W2Ff, 256, 128, 8);
    pack_frag_kernel<<<128, 256, 0, stream>>>(Wr2, W2Fr, 256, 128, 8);
    pack_frag_kernel<<<128, 256, 0, stream>>>(Wm2, W2Fm, 256, 128, 8);
    pack_frag_kernel<<<192, 256, 0, stream>>>(Wc, WCF, 384, 128, 8);

    // 3) fused per-stream MLP + segmented aggregation (all 3 streams, 1 launch)
    MlpArgs margs;
    margs.st[0] = {t_f, f_in, seg_f, W1Ff, bf1, W2Ff, bf2, Sf, Cf};
    margs.st[1] = {t_r, r_in, seg_r, W1Fr, br1, W2Fr, br2, Sr, Cr};
    margs.st[2] = {t_m, m_in, seg_m, W1Fm, bm1, W2Fm, bm2, Sm, Cm};
    margs.Wt = Wt;
    margs.bt = bt;
    const size_t smem1 = 135680;   // 67584 (X/Y) + 67584 (H) + 512 (seg)
    mlp_aggr_kernel<<<3 * NBLK, 256, smem1, stream>>>(margs);

    // 4) combine
    const size_t smem2 = 128 * CSTR * sizeof(_Float16);   // 100352
    combine_kernel<<<N_NODES / 128, 256, smem2, stream>>>(
        Sf, Sr, Sm, Cf, Cr, Cm, WCF, bc, (float*)d_out);
}
